// EmbedderGNN_6399501271360
// MI455X (gfx1250) — compile-verified, matrix operand order checked by reading, not run on hardware
//
#include <hip/hip_runtime.h>

#define DD 128

typedef __attribute__((ext_vector_type(16))) __bf16 v16bf;
typedef __attribute__((ext_vector_type(8)))  __bf16 v8bf;
typedef __attribute__((ext_vector_type(8)))  float  v8f;

// ---------------------------------------------------------------- utilities
__global__ void zero_f32(float* __restrict__ p, size_t n) {
  size_t i = (size_t)blockIdx.x * blockDim.x + threadIdx.x;
  const size_t stride = (size_t)gridDim.x * blockDim.x;
  for (; i < n; i += stride) p[i] = 0.0f;
}

// deg[dst] += 1 per edge (edge_index is int64, flat [2,E]: src row then dst row)
__global__ void degree_kernel(const long long* __restrict__ ei,
                              float* __restrict__ deg, int E) {
  int e = blockIdx.x * blockDim.x + threadIdx.x;
  if (e < E) atomicAdd(&deg[(int)ei[(size_t)E + e]], 1.0f);
}

// Pack Wcat[n][k] (k<128 from W_l, else W_r) as bf16, row-major [128][256].
__global__ void pack_weights(const float* __restrict__ Wl,
                             const float* __restrict__ Wr,
                             __bf16* __restrict__ Wcat) {
  int t = blockIdx.x * blockDim.x + threadIdx.x;   // exactly 128*256 threads
  int n = t >> 8, k = t & 255;
  float v = (k < DD) ? Wl[n * DD + k] : Wr[n * DD + (k - DD)];
  Wcat[t] = (__bf16)v;
}

// agg[dst] += feat[src]; one wave (32 lanes) per edge, 4 floats per lane.
__global__ void scatter_add(const long long* __restrict__ ei,
                            const float* __restrict__ feat,
                            float* __restrict__ agg, int E) {
  unsigned t = blockIdx.x * blockDim.x + threadIdx.x;
  unsigned e = t >> 5;
  if (e >= (unsigned)E) return;
  const unsigned lane = t & 31u;
  const int src = (int)ei[e];
  const int dst = (int)ei[(size_t)E + e];
  const float4 v = *(const float4*)(feat + (size_t)src * DD + lane * 4);
  float* a = agg + (size_t)dst * DD + lane * 4;
  atomicAdd(a + 0, v.x);
  atomicAdd(a + 1, v.y);
  atomicAdd(a + 2, v.z);
  atomicAdd(a + 3, v.w);
}

// ------------------------------------------------------- fused SAGE + LN
// out_tile = (agg/deg) @ W_l^T + hx @ W_r^T + bias       (WMMA bf16, K=256)
// out      = [relu]( LN(out_tile)*gamma+beta ) + res
// Block: 256 threads = 8 waves; block owns rows [16*bid, 16*bid+16), all 128 cols.
// Wave w computes the 16x16 column tile at n0 = 16*w.
__global__ __launch_bounds__(256)
void sage_ln_kernel(const float* __restrict__ agg, const float* __restrict__ hx,
                    const float* __restrict__ deg, const __bf16* __restrict__ Wcat,
                    const float* __restrict__ bias, const float* __restrict__ gamma,
                    const float* __restrict__ beta, const float* __restrict__ res,
                    float* __restrict__ out, int N, int do_relu) {
  __shared__ float tile[16][DD + 8];
  __shared__ float mu_s[16], rs_s[16];

  const int t    = threadIdx.x;
  const int wave = t >> 5;                 // 0..7 -> column tile
  const int lane = t & 31;
  const int half = lane >> 4;              // 0 or 1
  const int l16  = lane & 15;
  const int m0   = blockIdx.x * 16;
  const int n0   = wave * 16;

  // A-fragment source row for this lane (clamped so all 32 lanes stay active
  // for WMMA even on a ragged tail; stores are guarded below).
  int row = m0 + l16;
  if (row >= N) row = N - 1;
  const float invd = 1.0f / fmaxf(deg[row], 1.0f);
  const float*  arow = agg + (size_t)row * DD;
  const float*  hrow = hx  + (size_t)row * DD;
  const __bf16* wrow = Wcat + (size_t)(n0 + l16) * 256;   // B column n0+l16

  v8f acc = {};
  #pragma unroll
  for (int kb = 0; kb < 8; ++kb) {                 // K = 256 in steps of 32
    const int kbase = kb * 32 + half * 8;
    v16bf a, b;
    // 16-bit A 16x32 layout: lane (M=l16), elems 0..7 -> K=kbase..+7,
    //                                     elems 8..15 -> K=kbase+16..+23
    if (kb < 4) {                                  // first 128 K: mean-agg part
      #pragma unroll
      for (int i = 0; i < 8; ++i) {
        a[i]     = (__bf16)(arow[kbase + i]      * invd);
        a[8 + i] = (__bf16)(arow[kbase + 16 + i] * invd);
      }
    } else {                                       // last 128 K: self (lin_r) part
      const int k0 = kbase - DD;
      #pragma unroll
      for (int i = 0; i < 8; ++i) {
        a[i]     = (__bf16)hrow[k0 + i];
        a[8 + i] = (__bf16)hrow[k0 + 16 + i];
      }
    }
    // B 32x16: lane = column, same two 8-wide K runs; Wcat rows are contiguous.
    const v8bf blo = *(const v8bf*)(wrow + kbase);
    const v8bf bhi = *(const v8bf*)(wrow + kbase + 16);
    #pragma unroll
    for (int i = 0; i < 8; ++i) { b[i] = blo[i]; b[8 + i] = bhi[i]; }

    acc = __builtin_amdgcn_wmma_f32_16x16x32_bf16(
        /*neg_a=*/false, a, /*neg_b=*/false, b,
        /*c_mod=*/(short)0, acc, /*reuse_a=*/false, /*reuse_b=*/false);
  }

  // C/D layout: VGPR v -> row v + 8*half, col = l16. Add bias and park in LDS.
  const float bv = bias[n0 + l16];
  #pragma unroll
  for (int v = 0; v < 8; ++v)
    tile[half * 8 + v][n0 + l16] = acc[v] + bv;

  __syncthreads();

  // LayerNorm stats: one thread per row (rows are 128 wide, L2/LDS resident).
  if (t < 16) {
    float s = 0.0f;
    #pragma unroll 4
    for (int c = 0; c < DD; ++c) s += tile[t][c];
    const float mu = s * (1.0f / DD);
    float var = 0.0f;
    #pragma unroll 4
    for (int c = 0; c < DD; ++c) { const float d = tile[t][c] - mu; var += d * d; }
    mu_s[t] = mu;
    rs_s[t] = rsqrtf(var * (1.0f / DD) + 1e-5f);
  }
  __syncthreads();

  // Apply LN * gamma + beta, optional relu, residual; 8 elements per thread.
  const int r  = t >> 4;          // 0..15
  const int cb = (t & 15) * 8;    // 0..120
  const int gr = m0 + r;
  if (gr < N) {
    const float mu = mu_s[r], rs = rs_s[r];
    const float* resrow = res + (size_t)gr * DD;
    float*       outrow = out + (size_t)gr * DD;
    #pragma unroll
    for (int i = 0; i < 8; ++i) {
      const int c = cb + i;
      float val = (tile[r][c] - mu) * rs * gamma[c] + beta[c];
      if (do_relu) val = fmaxf(val, 0.0f);
      outrow[c] = val + resrow[c];
    }
  }
}

// ---------------------------------------------------------------- launcher
extern "C" void kernel_launch(void* const* d_in, const int* in_sizes, int n_in,
                              void* d_out, int out_size, void* d_ws, size_t ws_size,
                              hipStream_t stream) {
  const float*     x   = (const float*)d_in[0];
  const long long* ei  = (const long long*)d_in[1];   // int64 [2,E] flat
  const float*     W1l = (const float*)d_in[2];
  const float*     b1l = (const float*)d_in[3];
  const float*     W1r = (const float*)d_in[4];
  const float*     W2l = (const float*)d_in[5];
  const float*     b2l = (const float*)d_in[6];
  const float*     W2r = (const float*)d_in[7];
  const float*     g1  = (const float*)d_in[8];
  const float*     be1 = (const float*)d_in[9];
  const float*     g2  = (const float*)d_in[10];
  const float*     be2 = (const float*)d_in[11];

  const int N = in_sizes[0] / DD;
  const int E = in_sizes[1] / 2;

  // Workspace layout (fp32 unless noted):
  //   deg[N] | agg[N*D] | h[N*D] | Wcat1 bf16[128*256] | Wcat2 bf16[128*256]
  float*  ws  = (float*)d_ws;
  float*  deg = ws;
  float*  agg = ws + N;
  float*  h   = agg + (size_t)N * DD;
  __bf16* Wc1 = (__bf16*)(h + (size_t)N * DD);
  __bf16* Wc2 = Wc1 + 128 * 256;

  const int      ethreads    = (E + 255) / 256;
  const unsigned scat_blocks = (unsigned)(((size_t)E * 32 + 255) / 256);
  const int      mtiles      = (N + 15) / 16;

  // Degrees (shared by both layers) + packed bf16 weights.
  zero_f32<<<2048, 256, 0, stream>>>(deg, (size_t)N + (size_t)N * DD);
  degree_kernel<<<ethreads, 256, 0, stream>>>(ei, deg, E);
  pack_weights<<<128, 256, 0, stream>>>(W1l, W1r, Wc1);
  pack_weights<<<128, 256, 0, stream>>>(W2l, W2r, Wc2);

  // Layer 1: h = relu(LN(sage(x))) + x
  scatter_add<<<scat_blocks, 256, 0, stream>>>(ei, x, agg, E);
  sage_ln_kernel<<<mtiles, 256, 0, stream>>>(agg, x, deg, Wc1, b1l, g1, be1,
                                             x, h, N, /*relu=*/1);

  // Layer 2: out = LN(sage(h)) + h
  zero_f32<<<2048, 256, 0, stream>>>(agg, (size_t)N * DD);
  scatter_add<<<scat_blocks, 256, 0, stream>>>(ei, h, agg, E);
  sage_ln_kernel<<<mtiles, 256, 0, stream>>>(agg, h, deg, Wc2, b2l, g2, be2,
                                             h, (float*)d_out, N, /*relu=*/0);
}
